// NMT_20624432955546
// MI455X (gfx1250) — matrix operand order repaired and checked by reading, not compiled, not run on hardware
//
#include <hip/hip_runtime.h>
#include <hip/hip_bf16.h>

typedef __attribute__((ext_vector_type(16))) _Float16 v16h;
typedef __attribute__((ext_vector_type(8)))  _Float16 v8h;
typedef __attribute__((ext_vector_type(4)))  _Float16 v4h;
typedef __attribute__((ext_vector_type(8)))  float    v8f;

#define S_  64
#define T_  64
#define B_  128
#define H_  1024
#define H2_ 2048
#define H3_ 3072
#define K_  1024          // all GEMMs in this model have K = H = 1024
#define NCH 32            // K_/32 k-chunks
#define KG_ 16            // K_/64 staged k-groups

// ---------------------------------------------------------------------------
// WMMA GEMM, K fixed at 1024, register-blocked over M:
//   C[M,N] = A[M,K](f32) * Bpack(f16, WMMA-tile order) (+ bias) (+ beta*C)
// Block = 8 waves = 64(M) x 128(N). Each wave owns 1 N-tile x 4 M-tiles, so
// every B fragment (1KB coalesced load) is reused 4x from registers.
// A staged in LDS in 64x64 k-groups (8KB) with register software pipelining.
// ---------------------------------------------------------------------------
__global__ __launch_bounds__(256) void k_wmma_gemm(
    const float* __restrict__ A, const _Float16* __restrict__ Bp,
    const float* __restrict__ bias, float* __restrict__ C,
    int M, int N, int beta)
{
    __shared__ _Float16 As[64 * 64];   // 8 KB: 64 rows x 64 k (current group)

    const int tid  = threadIdx.x;
    const int lane = tid & 31;
    const int wave = tid >> 5;
    const int mB   = blockIdx.y << 6;                  // 64-row block base
    const int n0   = (blockIdx.x << 7) + (wave << 4);
    const int l16  = lane & 15;
    const int hi   = lane >> 4;
    const int ncol = n0 + l16;
    const int kA   = hi * 16;                          // contiguous k = 16h + s (matches Bpack)

    // ---- accumulators: 4 M-tiles ----
    v8f c[4];
#pragma unroll
    for (int mt = 0; mt < 4; ++mt) {
#pragma unroll
        for (int v = 0; v < 8; ++v) {
            const int m = mB + mt * 16 + v + hi * 8;
            float init = beta ? C[(size_t)m * N + ncol] : 0.0f;
            if (bias) init += bias[ncol];
            c[mt][v] = init;
        }
    }

    const _Float16* Bw = Bp + ((size_t)(n0 >> 4) * NCH) * 512 + (size_t)lane * 16;

    // ---- software pipeline: load k-group 0 into registers ----
    float4 r[4];
#pragma unroll
    for (int it = 0; it < 4; ++it) {
        const int idx = it * 256 + tid;                // over 64 rows x 16 f4
        r[it] = *(const float4*)(A + (size_t)(mB + (idx >> 4)) * K_ + (idx & 15) * 4);
    }

    for (int g = 0; g < KG_; ++g) {
        __syncthreads();                               // prior compute done
        // store staged registers -> LDS (f32 -> f16)
#pragma unroll
        for (int it = 0; it < 4; ++it) {
            const int idx = it * 256 + tid;
            v4h h4;
            h4[0] = (_Float16)r[it].x; h4[1] = (_Float16)r[it].y;
            h4[2] = (_Float16)r[it].z; h4[3] = (_Float16)r[it].w;
            *(v4h*)(As + (idx >> 4) * 64 + (idx & 15) * 4) = h4;
        }
        __syncthreads();

        // issue global loads for next k-group (overlaps with WMMA below)
        if (g + 1 < KG_) {
#pragma unroll
            for (int it = 0; it < 4; ++it) {
                const int idx = it * 256 + tid;
                r[it] = *(const float4*)(A + (size_t)(mB + (idx >> 4)) * K_
                                           + (g + 1) * 64 + (idx & 15) * 4);
            }
        }

        // compute: 2 k-chunks x 4 M-tiles = 8 WMMAs, B reused 4x per load
#pragma unroll
        for (int kc2 = 0; kc2 < 2; ++kc2) {
            const int chunk = g * 2 + kc2;
            v16h b = *(const v16h*)(Bw + (size_t)chunk * 512);
            if (chunk + 1 < NCH)
                __builtin_prefetch((const void*)(Bw + (size_t)(chunk + 1) * 512), 0, 1);
#pragma unroll
            for (int mt = 0; mt < 4; ++mt) {
                const _Float16* Ar = As + (mt * 16 + l16) * 64 + kc2 * 32 + kA;
                v8h a0 = *(const v8h*)Ar;
                v8h a1 = *(const v8h*)(Ar + 8);
                v16h a = __builtin_shufflevector(a0, a1,
                            0, 1, 2, 3, 4, 5, 6, 7, 8, 9, 10, 11, 12, 13, 14, 15);
                c[mt] = __builtin_amdgcn_wmma_f32_16x16x32_f16(
                            false, a, false, b, (short)0, c[mt], false, false);
            }
        }
    }

    // ---- epilogue ----
#pragma unroll
    for (int mt = 0; mt < 4; ++mt) {
#pragma unroll
        for (int v = 0; v < 8; ++v) {
            const int m = mB + mt * 16 + v + hi * 8;
            C[(size_t)m * N + ncol] = c[mt][v];
        }
    }
}

// ---------------------------------------------------------------------------
// Pack f32 weight [K_, N] into f16 WMMA tile order:
//   dst[((ntile*NCH + kc)*512) + lane*16 + i] = src[k, n]
//   n = ntile*16 + (lane&15);  k = kc*32 + (lane>>4)*16 + i
// ---------------------------------------------------------------------------
__global__ __launch_bounds__(256) void k_packB(
    const float* __restrict__ src, _Float16* __restrict__ dst, int N)
{
    size_t idx   = (size_t)blockIdx.x * 256 + threadIdx.x;   // over N*K_
    int tile     = (int)(idx >> 9);
    int within   = (int)(idx & 511);
    int lane     = within >> 4;
    int i        = within & 15;
    int ntile    = tile >> 5;          // / NCH
    int kc       = tile & 31;
    int n        = (ntile << 4) + (lane & 15);
    int k        = (kc << 5) + ((lane >> 4) << 4) + i;
    dst[idx] = (_Float16)src[(size_t)k * N + n];
}

__global__ __launch_bounds__(256) void k_copy(
    const float* __restrict__ src, float* __restrict__ dst, int n)
{
    int idx = blockIdx.x * blockDim.x + threadIdx.x;
    if (idx < n) dst[idx] = src[idx];
}

// ye[b,h] = emb[ys[b], h]
__global__ __launch_bounds__(256) void k_embed(
    const int* __restrict__ ys, const float* __restrict__ emb,
    float* __restrict__ ye)
{
    int idx = blockIdx.x * blockDim.x + threadIdx.x;
    int b = idx / H_, h = idx % H_;
    int id = ys[b];
    ye[idx] = emb[(size_t)id * H_ + h];
}

__device__ __forceinline__ float sigf(float x) { return 1.0f / (1.0f + __expf(-x)); }

// GRU gate: out = m*((1-z)*h + z*tanh(xh + r*hh)) + (1-m)*h
__global__ __launch_bounds__(256) void k_gru(
    const float* __restrict__ xg, const float* __restrict__ hg,
    const float* __restrict__ h,  const float* __restrict__ ymask,
    float* __restrict__ outp)
{
    int idx = blockIdx.x * blockDim.x + threadIdx.x;
    int b = idx / H_, j = idx % H_;
    size_t base = (size_t)b * H3_;
    float xz = xg[base + j], xr = xg[base + H_ + j], xh = xg[base + 2*H_ + j];
    float hz = hg[base + j], hr = hg[base + H_ + j], hh = hg[base + 2*H_ + j];
    float z = sigf(xz + hz);
    float r = sigf(xr + hr);
    float hv = h[idx];
    float hn = (1.0f - z) * hv + z * tanhf(xh + r * hh);
    float m = ymask[b];
    outp[idx] = m * hn + (1.0f - m) * hv;
}

// ---------------------------------------------------------------------------
// Attention, wave32-parallel: block per batch element, 8 waves.
// ---------------------------------------------------------------------------
__global__ __launch_bounds__(256) void k_attn(
    const float* __restrict__ ss, const float* __restrict__ u,
    const float* __restrict__ xsh, const float* __restrict__ xmask,
    const float* __restrict__ a1w, const float* __restrict__ a1b,
    float* __restrict__ ctx)
{
    const int b    = blockIdx.x;
    const int lane = threadIdx.x & 31;
    const int wave = threadIdx.x >> 5;
    __shared__ float ealpha[S_];
    const float* ssb = ss + (size_t)b * H_;

    for (int s = wave; s < S_; s += 8) {
        const float* us = u + ((size_t)s * B_ + b) * H_;
        float p = 0.0f;
        for (int h = lane; h < H_; h += 32)
            p += a1w[h] * tanhf(ssb[h] + us[h]);
#pragma unroll
        for (int o = 16; o > 0; o >>= 1) p += __shfl_xor(p, o, 32);
        if (lane == 0) ealpha[s] = p + a1b[0];
    }
    __syncthreads();

    if (threadIdx.x < 32) {
        float e0 = ealpha[threadIdx.x];
        float e1 = ealpha[threadIdx.x + 32];
        float m = fmaxf(e0, e1);
#pragma unroll
        for (int o = 16; o > 0; o >>= 1) m = fmaxf(m, __shfl_xor(m, o, 32));
        float x0 = __expf(e0 - m) * xmask[threadIdx.x * B_ + b];
        float x1 = __expf(e1 - m) * xmask[(threadIdx.x + 32) * B_ + b];
        float sum = x0 + x1;
#pragma unroll
        for (int o = 16; o > 0; o >>= 1) sum += __shfl_xor(sum, o, 32);
        float inv = 1.0f / (sum + 1e-20f);
        ealpha[threadIdx.x]      = x0 * inv;
        ealpha[threadIdx.x + 32] = x1 * inv;
    }
    __syncthreads();

    for (int h = threadIdx.x; h < H_; h += 256) {
        float acc = 0.0f;
#pragma unroll 8
        for (int s = 0; s < S_; ++s)
            acc += ealpha[s] * xsh[((size_t)s * B_ + b) * H_ + h];
        ctx[(size_t)b * H_ + h] = acc;
    }
}

// att = sigmoid(gbuf)*a1 + (1-sigmoid(gbuf))*a2
__global__ __launch_bounds__(256) void k_gate(
    const float* __restrict__ gbuf, const float* __restrict__ a1,
    const float* __restrict__ a2, float* __restrict__ att)
{
    int idx = blockIdx.x * blockDim.x + threadIdx.x;
    float g = sigf(gbuf[idx]);
    att[idx] = g * a1[idx] + (1.0f - g) * a2[idx];
}

// out[b,h] = max(l[b,2h], l[b,2h+1]) * ymask[b]
__global__ __launch_bounds__(256) void k_maxpair(
    const float* __restrict__ l, const float* __restrict__ ymask,
    float* __restrict__ outp)
{
    int idx = blockIdx.x * blockDim.x + threadIdx.x;
    int b = idx / H_, h = idx % H_;
    float v = fmaxf(l[(size_t)b * H2_ + 2*h], l[(size_t)b * H2_ + 2*h + 1]);
    outp[idx] = v * ymask[b];
}

// ---------------------------------------------------------------------------
extern "C" void kernel_launch(void* const* d_in, const int* in_sizes, int n_in,
                              void* d_out, int out_size, void* d_ws, size_t ws_size,
                              hipStream_t stream)
{
    const int*   ys      = (const int*)d_in[0];
    const float* xs_h    = (const float*)d_in[1];
    const float* uh      = (const float*)d_in[2];
    const float* xs_mask = (const float*)d_in[3];
    const float* ys_mask = (const float*)d_in[4];
    const float* s_tm1   = (const float*)d_in[5];
    const float* emb     = (const float*)d_in[6];
    const float* sa_w    = (const float*)d_in[7];
    const float* sa_b    = (const float*)d_in[8];
    const float* a1_w    = (const float*)d_in[9];
    const float* a1_b    = (const float*)d_in[10];
    const float* ha_w    = (const float*)d_in[11];
    const float* ha_b    = (const float*)d_in[12];
    const float* W1      = (const float*)d_in[13];
    const float* U1      = (const float*)d_in[14];
    const float* b1      = (const float*)d_in[15];
    const float* W2      = (const float*)d_in[16];
    const float* U2      = (const float*)d_in[17];
    const float* C2      = (const float*)d_in[18];
    const float* b2      = (const float*)d_in[19];
    const float* ua1_w   = (const float*)d_in[20];
    const float* ua1_b   = (const float*)d_in[21];
    const float* ua2_w   = (const float*)d_in[22];
    const float* ua2_b   = (const float*)d_in[23];
    const float* ls_w    = (const float*)d_in[24];
    const float* ls_b    = (const float*)d_in[25];
    const float* ly_w    = (const float*)d_in[26];
    const float* ly_b    = (const float*)d_in[27];
    const float* lc_w    = (const float*)d_in[28];
    const float* lc_b    = (const float*)d_in[29];
    float* out = (float*)d_out;

    char* ws = (char*)d_ws;
    size_t off = 0;
    auto alloc = [&](size_t bytes) -> char* {
        char* p = ws + off;
        off = (off + bytes + 255) & ~(size_t)255;
        return p;
    };

    // f16 packed weights (one-time conversion, WMMA tile order)
    _Float16* W1T  = (_Float16*)alloc((size_t)K_ * H3_ * 2);
    _Float16* U1T  = (_Float16*)alloc((size_t)K_ * H3_ * 2);
    _Float16* W2T  = (_Float16*)alloc((size_t)K_ * H3_ * 2);
    _Float16* U2T  = (_Float16*)alloc((size_t)K_ * H3_ * 2);
    _Float16* C2T  = (_Float16*)alloc((size_t)K_ * H3_ * 2);
    _Float16* sawT = (_Float16*)alloc((size_t)K_ * H_  * 2);
    _Float16* hawT = (_Float16*)alloc((size_t)K_ * H_  * 2);
    _Float16* ua1T = (_Float16*)alloc((size_t)K_ * H_  * 2);
    _Float16* ua2T = (_Float16*)alloc((size_t)K_ * H_  * 2);
    _Float16* lsT  = (_Float16*)alloc((size_t)K_ * H2_ * 2);
    _Float16* lyT  = (_Float16*)alloc((size_t)K_ * H2_ * 2);
    _Float16* lcT  = (_Float16*)alloc((size_t)K_ * H2_ * 2);
    // f32 activations / scratch
    float* btg    = (float*)alloc((size_t)S_ * B_ * H_ * 4);
    float* ye     = (float*)alloc((size_t)B_ * H_ * 4);
    float* s0     = (float*)alloc((size_t)B_ * H_ * 4);
    float* s1     = (float*)alloc((size_t)B_ * H_ * 4);
    float* s_ab   = (float*)alloc((size_t)B_ * H_ * 4);
    float* ssbuf  = (float*)alloc((size_t)B_ * H_ * 4);
    float* a1buf  = (float*)alloc((size_t)B_ * H_ * 4);
    float* a2buf  = (float*)alloc((size_t)B_ * H_ * 4);
    float* attbuf = (float*)alloc((size_t)B_ * H_ * 4);
    float* gbuf   = (float*)alloc((size_t)B_ * H_ * 4);
    float* xg     = (float*)alloc((size_t)B_ * H3_ * 4);
    float* hg     = (float*)alloc((size_t)B_ * H3_ * 4);
    float* lbuf   = (float*)alloc((size_t)B_ * H2_ * 4);

    auto pack = [&](const float* src, _Float16* dst, int N) {
        unsigned blocks = (unsigned)(((size_t)K_ * N) / 256);
        k_packB<<<blocks, 256, 0, stream>>>(src, dst, N);
    };
    auto gemm = [&](const float* A, const _Float16* Bp, const float* bias,
                    float* C, int M, int N, int beta) {
        k_wmma_gemm<<<dim3(N / 128, M / 64), 256, 0, stream>>>(
            A, Bp, bias, C, M, N, beta);
    };

    // ---- one-time precompute ----
    pack(W1, W1T, H3_);  pack(U1, U1T, H3_);
    pack(W2, W2T, H3_);  pack(U2, U2T, H3_);  pack(C2, C2T, H3_);
    pack(sa_w, sawT, H_); pack(ha_w, hawT, H_);
    pack(ua1_w, ua1T, H_); pack(ua2_w, ua2T, H_);
    pack(ls_w, lsT, H2_); pack(ly_w, lyT, H2_); pack(lc_w, lcT, H2_);

    // btg_uh = xs_h @ ha_w + ha_b   (8192 x 1024 x 1024 WMMA GEMM)
    gemm(xs_h, hawT, ha_b, btg, S_ * B_, H_, 0);

    k_copy<<<(B_ * H_) / 256, 256, 0, stream>>>(s_tm1, s0, B_ * H_);

    float* scur = s0;
    float* snext = s1;
    const int ew = (B_ * H_) / 256;

    for (int t = 0; t < T_; ++t) {
        const int*   yst = ys + t * B_;
        const float* ymt = ys_mask + t * B_;

        k_embed<<<ew, 256, 0, stream>>>(yst, emb, ye);

        // GRU 1
        gemm(ye,   W1T, b1,      xg, B_, H3_, 0);
        gemm(scur, U1T, nullptr, hg, B_, H3_, 0);
        k_gru<<<ew, 256, 0, stream>>>(xg, hg, scur, ymt, s_ab);

        // attention (shared s_above @ sa_w + sa_b)
        gemm(s_ab, sawT, sa_b, ssbuf, B_, H_, 0);
        k_attn<<<B_, 256, 0, stream>>>(ssbuf, uh,  xs_h, xs_mask, a1_w, a1_b, a1buf);
        k_attn<<<B_, 256, 0, stream>>>(ssbuf, btg, xs_h, xs_mask, a1_w, a1_b, a2buf);

        // GRU 2 (x = a1, h = s_above, c = a2)
        gemm(a1buf, W2T, b2,      xg, B_, H3_, 0);
        gemm(a2buf, C2T, nullptr, xg, B_, H3_, 1);
        gemm(s_ab,  U2T, nullptr, hg, B_, H3_, 0);
        k_gru<<<ew, 256, 0, stream>>>(xg, hg, s_ab, ymt, snext);

        // gate blend
        gemm(a1buf, ua1T, ua1_b, gbuf, B_, H_, 0);
        gemm(a2buf, ua2T, ua2_b, gbuf, B_, H_, 1);
        k_gate<<<ew, 256, 0, stream>>>(gbuf, a1buf, a2buf, attbuf);

        // logits + pairwise max
        gemm(snext,  lsT, ls_b, lbuf, B_, H2_, 0);
        gemm(ye,     lyT, ly_b, lbuf, B_, H2_, 1);
        gemm(attbuf, lcT, lc_b, lbuf, B_, H2_, 1);
        k_maxpair<<<ew, 256, 0, stream>>>(lbuf, ymt, out + (size_t)t * B_ * H_);

        float* tmp = scur; scur = snext; snext = tmp;
    }
}
